// GatNet_86363202388239
// MI455X (gfx1250) — matrix operand order repaired and checked by reading, not compiled, not run on hardware
//
#include <hip/hip_runtime.h>
#include <hip/hip_bf16.h>

typedef __bf16 bf16;
typedef __attribute__((ext_vector_type(16))) __bf16 v16bf;
typedef __attribute__((ext_vector_type(8)))  float  v8f;
typedef __attribute__((ext_vector_type(4)))  unsigned v4u;
typedef __attribute__((ext_vector_type(8)))  int      v8i_;
typedef __attribute__((ext_vector_type(4)))  int      v4i_;

#if defined(__has_builtin)
#if __has_builtin(__builtin_amdgcn_tensor_load_to_lds)
#define HAVE_TDM 1
#endif
#endif

namespace {
constexpr int N_NODES = 50000;
constexpr int N_EDGES = 800000;
constexpr int DIM_IN  = 512;
constexpr int DIM_R   = 128;
constexpr int DIM_F   = 64;
constexpr int N_HEADS = 4;
constexpr int DIM_OUT = 64;
constexpr int DIM_HF  = 256;   // H * F
constexpr float LRELU_ALPHA = 0.2f;
}

// ---------------------------------------------------------------------------
// fp32 -> bf16 cast (grid-stride)
// ---------------------------------------------------------------------------
__global__ void cvt_bf16_kernel(const float* __restrict__ in, bf16* __restrict__ out, int n) {
  for (int i = blockIdx.x * blockDim.x + threadIdx.x; i < n; i += gridDim.x * blockDim.x)
    out[i] = (bf16)in[i];
}

// ---------------------------------------------------------------------------
// Combined weight: Wc[g][f][k] = sum_r Wseq[g][f][r] * Wlin[g][r][k]  (-> bf16)
// ---------------------------------------------------------------------------
__global__ void combine_w_kernel(const float* __restrict__ Wseq,  // [G][F][R]
                                 const float* __restrict__ Wlin,  // [G][R][K]
                                 bf16* __restrict__ Wc,           // [G][F][K]
                                 int G, int Fdim, int Rdim, int K) {
  int idx = blockIdx.x * blockDim.x + threadIdx.x;
  if (idx >= G * Fdim * K) return;
  int k  = idx % K;
  int gf = idx / K;
  int f  = gf % Fdim;
  int g  = gf / Fdim;
  const float* ws = Wseq + ((size_t)g * Fdim + f) * Rdim;
  const float* wl = Wlin + (size_t)g * Rdim * K + k;
  float s = 0.f;
  for (int r = 0; r < Rdim; ++r) s += ws[r] * wl[(size_t)r * K];
  Wc[idx] = (bf16)s;
}

#ifdef HAVE_TDM
// ---------------------------------------------------------------------------
// Issue one 2-D TDM tile load: [tileY rows x tileX elems] of bf16 from a
// row-major tensor (row stride rowStride elems) into LDS at lds_addr, with
// hardware row padding: 16 dwords data + 4 dwords pad = 40 bf16 LDS row.
// Dims are given relative to the tile start so the M-tail zero-fills.
// ---------------------------------------------------------------------------
__device__ __forceinline__ void tdm_load_2d(unsigned lds_addr, const bf16* gaddr,
                                            int remX, int remY, int rowStride,
                                            int tileX, int tileY) {
  unsigned long long ga = (unsigned long long)(size_t)gaddr;
  v4u g0;
  g0[0] = 1u;                                   // count=1 (valid), user mode
  g0[1] = lds_addr;                             // lds_addr        [63:32]
  g0[2] = (unsigned)(ga & 0xffffffffull);       // global_addr lo  [95:64]
  g0[3] = (unsigned)((ga >> 32) & 0x1ffffffull) // global_addr hi  [120:96]
        | (2u << 30);                           // type=2 (image)  [127:126]
  v8i_ g1;
  g1[0] = (1 << 16)                             // data_size = 2 bytes
        | (1 << 20)                             // pad_enable
        | (3 << 22)                             // pad_interval: 16 dwords
        | (3 << 25);                            // pad_amount:    4 dwords
  g1[1] = (int)(((unsigned)remX & 0xffffu) << 16);             // dim0 lo [63:48]
  g1[2] = (int)(((unsigned)remX >> 16) |
                (((unsigned)remY & 0xffffu) << 16));           // dim0 hi | dim1 lo
  g1[3] = (int)(((unsigned)remY >> 16) |
                ((unsigned)tileX << 16));                      // dim1 hi | tile_dim0
  g1[4] = tileY;                                               // tile_dim1 (tile_dim2=0)
  g1[5] = rowStride;                                           // dim0_stride lo32
  g1[6] = 0;                                                   // dim0_stride hi | dim1_stride lo
  g1[7] = 0;
  v4i_ z4 = {0, 0, 0, 0};
  v8i_ z8 = {0, 0, 0, 0, 0, 0, 0, 0};
  __builtin_amdgcn_tensor_load_to_lds(g0, g1, z4, z4, z8, 0);
}
#endif

// ---------------------------------------------------------------------------
// C[g][M,64] = A[M,K](bf16) @ W[g][64,K](bf16)^T   using v_wmma_f32_16x16x32_bf16
// 256 threads = 8 waves; each wave -> 16 rows x 64 cols (4 WMMA accumulators).
// Tiles staged in LDS by the Tensor Data Mover (wave 0 issues, TENSORcnt
// tracked), double-buffered so the DMA overlaps the WMMAs.
// ---------------------------------------------------------------------------
__global__ __launch_bounds__(256) void gemm_bf16_n64(
    const bf16* __restrict__ A,     // [M,K] row-major
    const bf16* __restrict__ Wall,  // [gridDim.y][64][K] row-major
    float* __restrict__ Call,       // [gridDim.y][M][64]
    int M, int K)
{
  constexpr int TM = 128, TK = 32, PAD = 8, LDW = TK + PAD;
  __shared__ bf16 lA[2][TM][LDW];   // 2 x 128 x 40 bf16
  __shared__ bf16 lB[2][64][LDW];

  const bf16* W = Wall + (size_t)blockIdx.y * 64 * K;
  float*      C = Call + (size_t)blockIdx.y * M * 64;

  const int tid  = threadIdx.x;
  const int wave = tid >> 5;
  const int lane = tid & 31;
  const int col  = lane & 15;
  const int hi   = lane >> 4;
  const int rowStart = blockIdx.x * TM;
  const int rowBase  = rowStart + wave * 16;

  v8f acc[4] = {};

#ifdef HAVE_TDM
  if (wave == 0) {
    tdm_load_2d((unsigned)(size_t)&lA[0][0][0], A + (size_t)rowStart * K,
                K, M - rowStart, K, TK, TM);
    tdm_load_2d((unsigned)(size_t)&lB[0][0][0], W, K, 64, K, TK, 64);
  }
#endif

  int buf = 0;
  for (int k0 = 0; k0 < K; k0 += TK, buf ^= 1) {
#ifdef HAVE_TDM
    if (wave == 0) __builtin_amdgcn_s_wait_tensorcnt(0);
    __syncthreads();                       // tile `buf` published to all waves
    int k1 = k0 + TK;
    if (wave == 0 && k1 < K) {             // prefetch next tile into other buffer
      tdm_load_2d((unsigned)(size_t)&lA[buf ^ 1][0][0],
                  A + (size_t)rowStart * K + k1, K - k1, M - rowStart, K, TK, TM);
      tdm_load_2d((unsigned)(size_t)&lB[buf ^ 1][0][0],
                  W + k1, K - k1, 64, K, TK, 64);
    }
#else
    // manual staging fallback (no TDM builtin on this toolchain)
#pragma unroll
    for (int i = 0; i < 2; ++i) {
      int idx = tid + i * 256;
      int r   = idx >> 2;
      int cc  = (idx & 3) * 8;
      int gr  = rowStart + r;
      uint4 v = make_uint4(0u, 0u, 0u, 0u);
      if (gr < M) v = *(const uint4*)(A + (size_t)gr * K + k0 + cc);
      *(uint4*)(&lA[buf][r][cc]) = v;
    }
    {
      int r  = tid >> 2;
      int cc = (tid & 3) * 8;
      *(uint4*)(&lB[buf][r][cc]) = *(const uint4*)(W + (size_t)r * K + k0 + cc);
    }
    __syncthreads();
#endif

    // A fragment: lane holds row (lane&15); lanes 0-15: K=0..15;
    // lanes 16-31: K=16..31 (2 bf16 per dword).
    union { v16bf v; unsigned u[8]; } af;
    {
      const unsigned* ar = (const unsigned*)(&lA[buf][wave * 16 + col][0]);
#pragma unroll
      for (int j = 0; j < 4; ++j) {
        af.u[j]     = ar[hi * 8 + j];
        af.u[4 + j] = ar[hi * 8 + 4 + j];
      }
    }
    // 4 column-blocks of 16; B fragment: lane holds col (lane&15);
    // lanes 0-15: K=0..15, lanes 16-31: K=16..31.
#pragma unroll
    for (int nb = 0; nb < 4; ++nb) {
      union { v16bf v; unsigned u[8]; } bfr;
      const unsigned* br = (const unsigned*)(&lB[buf][nb * 16 + col][0]);
#pragma unroll
      for (int j = 0; j < 8; ++j) bfr.u[j] = br[hi * 8 + j];
      acc[nb] = __builtin_amdgcn_wmma_f32_16x16x32_bf16(
          false, af.v, false, bfr.v, (short)0, acc[nb], false, false);
    }
#ifndef HAVE_TDM
    __syncthreads();
#endif
  }

  // C/D layout: lane 0-15 = N, rows M=0..7 in VGPR0..7; lanes 16-31 rows 8..15.
#pragma unroll
  for (int nb = 0; nb < 4; ++nb) {
#pragma unroll
    for (int r = 0; r < 8; ++r) {
      int gr = rowBase + hi * 8 + r;
      if (gr < M) C[(size_t)gr * 64 + nb * 16 + col] = acc[nb][r];
    }
  }
}

// ---------------------------------------------------------------------------
// f1[h,n] = S[h,n,:].a1[h] + b1[h];  f2 likewise. One thread per (h,n).
// ---------------------------------------------------------------------------
__global__ void f1f2_kernel(const float* __restrict__ S,
                            const float* __restrict__ a1, const float* __restrict__ b1,
                            const float* __restrict__ a2, const float* __restrict__ b2,
                            float* __restrict__ f1, float* __restrict__ f2,
                            int n, int nh) {
  int idx = blockIdx.x * blockDim.x + threadIdx.x;
  if (idx >= n * nh) return;
  int h = idx / n;
  const float* s  = S + (size_t)idx * 64;
  const float* A1 = a1 + h * 64;
  const float* A2 = a2 + h * 64;
  float s1 = b1[h], s2 = b2[h];
#pragma unroll 8
  for (int f = 0; f < 64; ++f) {
    float v = s[f];
    s1 += v * A1[f];
    s2 += v * A2[f];
  }
  f1[idx] = s1;
  f2[idx] = s2;
}

// ---------------------------------------------------------------------------
// Edge pass: one wave per (edge, head); lane owns 2 of 64 features.
// coef = exp(leaky_relu(f1[src]+f2[dst])); denom[src]+=coef; agg[src]+=coef*S[dst].
// ---------------------------------------------------------------------------
__global__ __launch_bounds__(256) void edge_kernel(
    const int* __restrict__ src, const int* __restrict__ dst,
    const float* __restrict__ S, const float* __restrict__ f1,
    const float* __restrict__ f2, float* __restrict__ denom,
    float* __restrict__ agg, int nHeads)
{
  long long gw = ((long long)blockIdx.x * blockDim.x + threadIdx.x) >> 5;
  long long total = (long long)N_EDGES * nHeads;
  if (gw >= total) return;
  int lane = threadIdx.x & 31;
  int h = (int)(gw / N_EDGES);
  int e = (int)(gw - (long long)h * N_EDGES);
  int s = src[e], d = dst[e];
  float lg = f1[(size_t)h * N_NODES + s] + f2[(size_t)h * N_NODES + d];
  float lr = lg > 0.f ? lg : LRELU_ALPHA * lg;
  float cf = __expf(lr);
  if (lane == 0) atomicAdd(denom + (size_t)h * N_NODES + s, cf);
  const float2 sv = *(const float2*)(S + ((size_t)h * N_NODES + d) * 64 + lane * 2);
  float* ap = agg + ((size_t)h * N_NODES + s) * 64 + lane * 2;
  atomicAdd(ap,     cf * sv.x);
  atomicAdd(ap + 1, cf * sv.y);
}

// ---------------------------------------------------------------------------
// heads finalize: elu(agg/denom + bias) -> hcat bf16 [N, 256]
// ---------------------------------------------------------------------------
__global__ void finalize_heads_kernel(const float* __restrict__ agg,
                                      const float* __restrict__ denom,
                                      const float* __restrict__ bias,
                                      bf16* __restrict__ hcat) {
  int idx = blockIdx.x * blockDim.x + threadIdx.x;  // over H*N*64
  if (idx >= N_HEADS * N_NODES * DIM_F) return;
  int f  = idx & 63;
  int hn = idx >> 6;              // h*N + n
  int h  = hn / N_NODES;
  int n  = hn - h * N_NODES;
  float v = agg[idx] / denom[hn] + bias[h * DIM_F + f];
  float e = v > 0.f ? v : (__expf(v) - 1.f);
  hcat[(size_t)n * DIM_HF + h * DIM_F + f] = (bf16)e;
}

// ---------------------------------------------------------------------------
// classifier finalize: out = agg/denom + bias (no ELU)
// ---------------------------------------------------------------------------
__global__ void finalize_out_kernel(const float* __restrict__ agg,
                                    const float* __restrict__ denom,
                                    const float* __restrict__ bias,
                                    float* __restrict__ out) {
  int idx = blockIdx.x * blockDim.x + threadIdx.x;  // over N*64
  if (idx >= N_NODES * DIM_OUT) return;
  int f = idx & 63;
  int n = idx >> 6;
  out[idx] = agg[idx] / denom[n] + bias[f];
}

// ---------------------------------------------------------------------------
extern "C" void kernel_launch(void* const* d_in, const int* in_sizes, int n_in,
                              void* d_out, int out_size, void* d_ws, size_t ws_size,
                              hipStream_t stream) {
  (void)in_sizes; (void)n_in; (void)out_size; (void)ws_size;
  const float* x      = (const float*)d_in[0];
  const int*   edges  = (const int*)  d_in[1];
  const float* Wlin_h = (const float*)d_in[2];
  const float* Wseq_h = (const float*)d_in[3];
  const float* a1_h   = (const float*)d_in[4];
  const float* b1_h   = (const float*)d_in[5];
  const float* a2_h   = (const float*)d_in[6];
  const float* b2_h   = (const float*)d_in[7];
  const float* bias_h = (const float*)d_in[8];
  const float* Wlin_c = (const float*)d_in[9];
  const float* Wseq_c = (const float*)d_in[10];
  const float* a1_c   = (const float*)d_in[11];
  const float* b1_c   = (const float*)d_in[12];
  const float* a2_c   = (const float*)d_in[13];
  const float* b2_c   = (const float*)d_in[14];
  const float* bias_c = (const float*)d_in[15];
  float* out = (float*)d_out;
  const int* src = edges;
  const int* dst = edges + N_EDGES;

  char* p = (char*)d_ws;
  auto take = [&](size_t bytes) { char* r = p; p += (bytes + 255) & ~(size_t)255; return (void*)r; };
  bf16*  Xb   = (bf16*) take((size_t)N_NODES * DIM_IN * 2);
  bf16*  Wch  = (bf16*) take((size_t)N_HEADS * DIM_F * DIM_IN * 2);
  bf16*  Wcc  = (bf16*) take((size_t)DIM_OUT * DIM_HF * 2);
  float* S    = (float*)take((size_t)N_HEADS * N_NODES * DIM_F * 4);
  float* f1h  = (float*)take((size_t)N_HEADS * N_NODES * 4);
  float* f2h  = (float*)take((size_t)N_HEADS * N_NODES * 4);
  float* denh = (float*)take((size_t)N_HEADS * N_NODES * 4);
  float* aggh = (float*)take((size_t)N_HEADS * N_NODES * DIM_F * 4);
  bf16*  hcat = (bf16*) take((size_t)N_NODES * DIM_HF * 2);
  float* Sc   = (float*)take((size_t)N_NODES * DIM_OUT * 4);
  float* f1c  = (float*)take((size_t)N_NODES * 4);
  float* f2c  = (float*)take((size_t)N_NODES * 4);
  float* denc = (float*)take((size_t)N_NODES * 4);
  float* aggc = (float*)take((size_t)N_NODES * DIM_OUT * 4);

  (void)hipMemsetAsync(denh, 0, (size_t)N_HEADS * N_NODES * 4, stream);
  (void)hipMemsetAsync(aggh, 0, (size_t)N_HEADS * N_NODES * DIM_F * 4, stream);
  (void)hipMemsetAsync(denc, 0, (size_t)N_NODES * 4, stream);
  (void)hipMemsetAsync(aggc, 0, (size_t)N_NODES * DIM_OUT * 4, stream);

  // 1) x -> bf16
  cvt_bf16_kernel<<<2048, 256, 0, stream>>>(x, Xb, N_NODES * DIM_IN);

  // 2) combined weights (Ws @ Wl) per head, then classifier
  {
    int tot = N_HEADS * DIM_F * DIM_IN;
    combine_w_kernel<<<(tot + 255) / 256, 256, 0, stream>>>(
        Wseq_h, Wlin_h, Wch, N_HEADS, DIM_F, DIM_R, DIM_IN);
  }
  {
    int tot = DIM_OUT * DIM_HF;
    combine_w_kernel<<<(tot + 255) / 256, 256, 0, stream>>>(
        Wseq_c, Wlin_c, Wcc, 1, DIM_OUT, DIM_R, DIM_HF);
  }

  // 3) head GEMMs: S[h] = Xb @ Wch[h]^T   (WMMA bf16, TDM-staged tiles)
  {
    dim3 g((N_NODES + 127) / 128, N_HEADS);
    gemm_bf16_n64<<<g, 256, 0, stream>>>(Xb, Wch, S, N_NODES, DIM_IN);
  }

  // 4) f1/f2 per head
  {
    int tot = N_HEADS * N_NODES;
    f1f2_kernel<<<(tot + 255) / 256, 256, 0, stream>>>(
        S, a1_h, b1_h, a2_h, b2_h, f1h, f2h, N_NODES, N_HEADS);
  }

  // 5) edge attention pass (heads)
  {
    long long waves = (long long)N_EDGES * N_HEADS;
    long long blocks = (waves * 32 + 255) / 256;
    edge_kernel<<<(unsigned)blocks, 256, 0, stream>>>(src, dst, S, f1h, f2h, denh, aggh, N_HEADS);
  }

  // 6) finalize heads -> hcat (bf16)
  {
    int tot = N_HEADS * N_NODES * DIM_F;
    finalize_heads_kernel<<<(tot + 255) / 256, 256, 0, stream>>>(aggh, denh, bias_h, hcat);
  }

  // 7) classifier GEMM: Sc = hcat @ Wcc^T (WMMA bf16)
  {
    dim3 g((N_NODES + 127) / 128, 1);
    gemm_bf16_n64<<<g, 256, 0, stream>>>(hcat, Wcc, Sc, N_NODES, DIM_HF);
  }

  // 8) f1/f2 classifier
  {
    int tot = N_NODES;
    f1f2_kernel<<<(tot + 255) / 256, 256, 0, stream>>>(
        Sc, a1_c, b1_c, a2_c, b2_c, f1c, f2c, N_NODES, 1);
  }

  // 9) edge attention pass (classifier)
  {
    long long waves = (long long)N_EDGES;
    long long blocks = (waves * 32 + 255) / 256;
    edge_kernel<<<(unsigned)blocks, 256, 0, stream>>>(src, dst, Sc, f1c, f2c, denc, aggc, 1);
  }

  // 10) finalize -> d_out
  {
    int tot = N_NODES * DIM_OUT;
    finalize_out_kernel<<<(tot + 255) / 256, 256, 0, stream>>>(aggc, denc, bias_c, out);
  }
}
